// MambaBlock_16776142258506
// MI455X (gfx1250) — matrix operand order repaired and checked by reading, not compiled, not run on hardware
//
#include <hip/hip_runtime.h>

// ---------------- model dims ----------------
#define BB 4
#define LL 2048
#define DMODEL 1024
#define DSTATE 16
#define DCONV 4
#define DINNER 2048
#define DTRANK 64
#define NROWS (BB*LL)          // 8192

typedef __attribute__((ext_vector_type(16))) __bf16 v16bf;
typedef __attribute__((ext_vector_type(8)))  float  v8f;

__device__ __forceinline__ unsigned short f2bf(float f) {
  unsigned int u = __float_as_uint(f);
  unsigned int r = (u + 0x7FFFu + ((u >> 16) & 1u)) >> 16;
  return (unsigned short)r;
}

// async copy 16B global -> LDS (per-lane), tracked by ASYNCcnt
__device__ __forceinline__ void async_b128(unsigned ldsAddr, const void* gaddr) {
  asm volatile("global_load_async_to_lds_b128 %0, %1, off"
               :: "v"(ldsAddr), "v"(gaddr) : "memory");
}
__device__ __forceinline__ void wait_async0() {
  asm volatile("s_wait_asynccnt 0x0" ::: "memory");
}

// ---------------- elementwise: f32 -> bf16 ----------------
__global__ __launch_bounds__(256) void cvt_bf16(const float* __restrict__ in,
                                                unsigned short* __restrict__ out, int n) {
  int i = blockIdx.x * 256 + threadIdx.x;
  if (i < n) out[i] = f2bf(in[i]);
}

// ---------------- layernorm (D=1024) -> bf16 ----------------
__global__ __launch_bounds__(256) void ln_bf16(const float* __restrict__ x,
                                               const float* __restrict__ g,
                                               const float* __restrict__ b,
                                               unsigned short* __restrict__ out) {
  int row = blockIdx.x, tid = threadIdx.x;
  const float4 v = *(const float4*)(x + (size_t)row * DMODEL + tid * 4);
  float s  = v.x + v.y + v.z + v.w;
  float sq = v.x*v.x + v.y*v.y + v.z*v.z + v.w*v.w;
  for (int off = 16; off > 0; off >>= 1) {
    s  += __shfl_down(s,  off, 32);
    sq += __shfl_down(sq, off, 32);
  }
  __shared__ float ss[8], ssq[8];
  int lane = tid & 31, w = tid >> 5;
  if (lane == 0) { ss[w] = s; ssq[w] = sq; }
  __syncthreads();
  if (tid == 0) {
    float a = 0.f, c = 0.f;
    for (int i = 0; i < 8; ++i) { a += ss[i]; c += ssq[i]; }
    ss[0] = a; ssq[0] = c;
  }
  __syncthreads();
  float mu   = ss[0]  * (1.0f / DMODEL);
  float var  = ssq[0] * (1.0f / DMODEL) - mu * mu;
  float rstd = rsqrtf(var + 1e-5f);
  int c0 = tid * 4;
  float vv[4] = {v.x, v.y, v.z, v.w};
  for (int i = 0; i < 4; ++i) {
    float o = (vv[i] - mu) * rstd * g[c0 + i] + b[c0 + i];
    out[(size_t)row * DMODEL + c0 + i] = f2bf(o);
  }
}

// ---------------- bf16 WMMA GEMM: out[M,N] = A[M,lda..K] * W[N,K]^T ----------------
constexpr int EP_F32 = 0, EP_F32_BF16 = 1, EP_SOFTPLUS = 2,
              EP_BIAS_RELU_BF16 = 3, EP_ADDRES = 4, EP_BIAS_ADD = 5;
#define LDT 40  // LDS row stride (elements) to dodge bank conflicts, 16B-aligned

template <int EP>
__global__ __launch_bounds__(128)
void gemm_bf16(const unsigned short* __restrict__ A, int lda,
               const unsigned short* __restrict__ W,
               const float* __restrict__ bias,
               const float* __restrict__ res,
               float* __restrict__ outf,
               unsigned short* __restrict__ outbf,
               int M, int N, int K) {
  __shared__ unsigned short As[2][64 * LDT];
  __shared__ unsigned short Ws[2][64 * LDT];
  int tid  = threadIdx.x;
  int lane = tid & 31, wave = tid >> 5;
  int half = lane >> 4, r = lane & 15;
  int wm = (wave >> 1) * 32, wn = (wave & 1) * 32;
  int tileM = blockIdx.x * 64, tileN = blockIdx.y * 64;

  // this thread's two 16B copy chunks per 64x32 tile
  int cid0 = tid,        row0 = cid0 >> 2, c80 = (cid0 & 3) * 8;
  int cid1 = tid + 128,  row1 = cid1 >> 2, c81 = (cid1 & 3) * 8;
  const unsigned short* aG0 = A + (size_t)(tileM + row0) * lda + c80;
  const unsigned short* aG1 = A + (size_t)(tileM + row1) * lda + c81;
  int wr0 = (tileN + row0 < N) ? tileN + row0 : N - 1;  // clamp; cols >= N never stored
  int wr1 = (tileN + row1 < N) ? tileN + row1 : N - 1;
  const unsigned short* wG0 = W + (size_t)wr0 * K + c80;
  const unsigned short* wG1 = W + (size_t)wr1 * K + c81;
  unsigned aL0[2], aL1[2], wL0[2], wL1[2];
  for (int bf = 0; bf < 2; ++bf) {
    aL0[bf] = (unsigned)(unsigned long long)&As[bf][row0 * LDT + c80];
    aL1[bf] = (unsigned)(unsigned long long)&As[bf][row1 * LDT + c81];
    wL0[bf] = (unsigned)(unsigned long long)&Ws[bf][row0 * LDT + c80];
    wL1[bf] = (unsigned)(unsigned long long)&Ws[bf][row1 * LDT + c81];
  }

  v8f acc[2][2] = {};
  int kTiles = K >> 5;

  // preload k-tile 0 into buffer 0
  async_b128(aL0[0], aG0);
  async_b128(aL1[0], aG1);
  async_b128(wL0[0], wG0);
  async_b128(wL1[0], wG1);
  wait_async0();
  __syncthreads();

  for (int kt = 0; kt < kTiles; ++kt) {
    int buf = kt & 1;
    if (kt + 1 < kTiles) {  // prefetch next tile into the other buffer (uniform branch)
      int ko = (kt + 1) * 32;
      async_b128(aL0[buf ^ 1], aG0 + ko);
      async_b128(aL1[buf ^ 1], aG1 + ko);
      async_b128(wL0[buf ^ 1], wG0 + ko);
      async_b128(wL1[buf ^ 1], wG1 + ko);
    }
    union Frag { v16bf v; uint4 u[2]; };
    Frag a[2], bfr[2];
    for (int i = 0; i < 2; ++i) {  // A 16x32: lane half h reads K [8h,8h+8) and [16+8h,24+8h)
      const unsigned short* ap = &As[buf][(wm + i * 16 + r) * LDT + half * 8];
      a[i].u[0] = *(const uint4*)ap;
      a[i].u[1] = *(const uint4*)(ap + 16);
    }
    for (int j = 0; j < 2; ++j) {  // B 32x16: lane half h reads K [8h,8h+8) and [16+8h,24+8h)
      const unsigned short* bp = &Ws[buf][(wn + j * 16 + r) * LDT + half * 8];
      bfr[j].u[0] = *(const uint4*)bp;
      bfr[j].u[1] = *(const uint4*)(bp + 16);
    }
    for (int i = 0; i < 2; ++i)
      for (int j = 0; j < 2; ++j)
        acc[i][j] = __builtin_amdgcn_wmma_f32_16x16x32_bf16(
            false, a[i].v, false, bfr[j].v, (short)0, acc[i][j], false, false);
    wait_async0();
    __syncthreads();
  }

  // epilogue: D element (vgpr rr, lane) -> row half*8+rr, col lane&15
  for (int i = 0; i < 2; ++i)
    for (int j = 0; j < 2; ++j) {
      int col = tileN + wn + j * 16 + r;
      if (col >= N) continue;
      float bv = 0.f;
      if (EP == EP_SOFTPLUS || EP == EP_BIAS_RELU_BF16 || EP == EP_BIAS_ADD) bv = bias[col];
      for (int rr = 0; rr < 8; ++rr) {
        int row = tileM + wm + i * 16 + half * 8 + rr;
        float v = acc[i][j][rr];
        size_t o = (size_t)row * N + col;
        if (EP == EP_F32) {
          outf[o] = v;
        } else if (EP == EP_F32_BF16) {
          outf[o] = v; outbf[o] = f2bf(v);
        } else if (EP == EP_SOFTPLUS) {
          v += bv; outf[o] = (v > 20.f) ? v : log1pf(__expf(v));
        } else if (EP == EP_BIAS_RELU_BF16) {
          v += bv; outbf[o] = f2bf(fmaxf(v, 0.f));
        } else if (EP == EP_ADDRES) {
          outf[o] = v + res[o];
        } else if (EP == EP_BIAS_ADD) {
          outf[o] = v + bv + res[o];
        }
      }
    }
}

// ---------------- causal depthwise conv (K=4) + bias + SiLU ----------------
__global__ __launch_bounds__(256) void conv_silu(const float* __restrict__ xz,
                                                 const float* __restrict__ cw,
                                                 const float* __restrict__ cb,
                                                 float* __restrict__ xi,
                                                 unsigned short* __restrict__ xibf) {
  int idx = blockIdx.x * 256 + threadIdx.x;  // over B*L*DINNER
  int d  = idx & (DINNER - 1);
  int bl = idx >> 11;
  int l  = bl & (LL - 1);
  float w0 = cw[d * 4 + 0], w1 = cw[d * 4 + 1], w2 = cw[d * 4 + 2], w3 = cw[d * 4 + 3];
  const float* base = xz + (size_t)bl * (2 * DINNER) + d;
  float v = cb[d] + w3 * base[0];
  if (l >= 1) v += w2 * base[-(2 * DINNER)];
  if (l >= 2) v += w1 * base[-2 * (2 * DINNER)];
  if (l >= 3) v += w0 * base[-3 * (2 * DINNER)];
  v = v / (1.f + __expf(-v));  // silu
  xi[idx] = v;
  xibf[idx] = f2bf(v);
}

// ---------------- sequential SSM scan: 1 lane per (b, channel) ----------------
__global__ __launch_bounds__(256) void ssm_scan(const float* __restrict__ xdbl,  // [B*L,96]
                                                float* __restrict__ dt_ys,       // in: dt, out: ys
                                                const float* __restrict__ xi,
                                                const float* __restrict__ A_log) {
  int tid = threadIdx.x;
  int b = blockIdx.x >> 3;
  int d = ((blockIdx.x & 7) << 8) + tid;
  float Arow[DSTATE];
#pragma unroll
  for (int n = 0; n < DSTATE; ++n) Arow[n] = -__expf(A_log[d * DSTATE + n]);
  float s[DSTATE] = {};
  __shared__ float sB[64][DSTATE], sC[64][DSTATE];
  for (int t0 = 0; t0 < LL; t0 += 64) {
    __syncthreads();
    for (int p = 0; p < 8; ++p) {
      int idx = tid + (p << 8);
      int tt = idx >> 5, q = idx & 31;
      float v = xdbl[(size_t)(b * LL + t0 + tt) * 96 + DTRANK + q];
      if (q < 16) sB[tt][q] = v; else sC[tt][q - 16] = v;
    }
    __syncthreads();
    for (int tt = 0; tt < 64; ++tt) {
      size_t o = (size_t)(b * LL + t0 + tt) * DINNER + d;
      float dtv = dt_ys[o];
      float u = xi[o];
      float dtu = dtv * u;
      float y = 0.f;
#pragma unroll
      for (int n = 0; n < DSTATE; ++n) {
        s[n] = __expf(dtv * Arow[n]) * s[n] + sB[tt][n] * dtu;
        y += s[n] * sC[tt][n];
      }
      dt_ys[o] = y;  // in-place: dt consumed before overwrite
    }
  }
}

// ---------------- gate: (ys + Dp*xi) * silu(z) -> bf16 ----------------
__global__ __launch_bounds__(256) void gate_silu(const float* __restrict__ ys,
                                                 const float* __restrict__ xi,
                                                 const float* __restrict__ xz,
                                                 const float* __restrict__ Dp,
                                                 unsigned short* __restrict__ ybf) {
  int idx = blockIdx.x * 256 + threadIdx.x;
  int d  = idx & (DINNER - 1);
  int bl = idx >> 11;
  float z = xz[(size_t)bl * (2 * DINNER) + DINNER + d];
  float y = (ys[idx] + Dp[d] * xi[idx]) * (z / (1.f + __expf(-z)));
  ybf[idx] = f2bf(y);
}

// ---------------- host launch ----------------
static inline size_t al256(size_t x) { return (x + 255) & ~(size_t)255; }

extern "C" void kernel_launch(void* const* d_in, const int* in_sizes, int n_in,
                              void* d_out, int out_size, void* d_ws, size_t ws_size,
                              hipStream_t stream) {
  const float* x        = (const float*)d_in[0];
  const float* ln1_g    = (const float*)d_in[1];
  const float* ln1_b    = (const float*)d_in[2];
  const float* ln2_g    = (const float*)d_in[3];
  const float* ln2_b    = (const float*)d_in[4];
  const float* in_proj  = (const float*)d_in[5];
  const float* conv_w   = (const float*)d_in[6];
  const float* conv_b   = (const float*)d_in[7];
  const float* x_proj   = (const float*)d_in[8];
  const float* dt_proj  = (const float*)d_in[9];
  const float* dt_b     = (const float*)d_in[10];
  const float* A_log    = (const float*)d_in[11];
  const float* Dp       = (const float*)d_in[12];
  const float* out_proj = (const float*)d_in[13];
  const float* ffn_w1   = (const float*)d_in[14];
  const float* ffn_b1   = (const float*)d_in[15];
  const float* ffn_w2   = (const float*)d_in[16];
  const float* ffn_b2   = (const float*)d_in[17];
  float* out = (float*)d_out;

  char* base = (char*)d_ws;
  size_t cur = 0;
  auto take = [&](size_t bytes) { char* p = base + cur; cur = al256(cur + bytes); return p; };

  unsigned short* w_in  = (unsigned short*)take((size_t)2 * DINNER * DMODEL * 2);   // [4096,1024]
  unsigned short* w_xp  = (unsigned short*)take((size_t)96 * DINNER * 2);           // [96,2048]
  unsigned short* w_dt  = (unsigned short*)take((size_t)DINNER * DTRANK * 2);       // [2048,64]
  unsigned short* w_op  = (unsigned short*)take((size_t)DMODEL * DINNER * 2);       // [1024,2048]
  unsigned short* w_f1  = (unsigned short*)take((size_t)4 * DMODEL * DMODEL * 2);   // [4096,1024]
  unsigned short* w_f2  = (unsigned short*)take((size_t)DMODEL * 4 * DMODEL * 2);   // [1024,4096]
  unsigned short* h1bf  = (unsigned short*)take((size_t)NROWS * DMODEL * 2);
  float*          xz    = (float*)take((size_t)NROWS * 2 * DINNER * 4);
  float*          xi    = (float*)take((size_t)NROWS * DINNER * 4);
  unsigned short* xibf  = (unsigned short*)take((size_t)NROWS * DINNER * 2);
  float*          xdbl  = (float*)take((size_t)NROWS * 96 * 4);
  unsigned short* xdblb = (unsigned short*)take((size_t)NROWS * 96 * 2);
  float*          dt_ys = (float*)take((size_t)NROWS * DINNER * 4);
  unsigned short* ybf   = (unsigned short*)take((size_t)NROWS * DINNER * 2);
  float*          x2    = (float*)take((size_t)NROWS * DMODEL * 4);
  unsigned short* h2bf  = (unsigned short*)take((size_t)NROWS * DMODEL * 2);
  unsigned short* f1bf  = (unsigned short*)take((size_t)NROWS * 4 * DMODEL * 2);
  (void)ws_size; (void)n_in; (void)in_sizes; (void)out_size;

  // weight conversions f32 -> bf16
  auto cvt = [&](const float* src, unsigned short* dst, int n) {
    cvt_bf16<<<(n + 255) / 256, 256, 0, stream>>>(src, dst, n);
  };
  cvt(in_proj,  w_in, 2 * DINNER * DMODEL);
  cvt(x_proj,   w_xp, 96 * DINNER);
  cvt(dt_proj,  w_dt, DINNER * DTRANK);
  cvt(out_proj, w_op, DMODEL * DINNER);
  cvt(ffn_w1,   w_f1, 4 * DMODEL * DMODEL);
  cvt(ffn_w2,   w_f2, DMODEL * 4 * DMODEL);

  // LN1 -> bf16
  ln_bf16<<<NROWS, 256, 0, stream>>>(x, ln1_g, ln1_b, h1bf);

  // in_proj: [8192,1024] x [4096,1024]^T -> xz f32
  gemm_bf16<EP_F32><<<dim3(NROWS / 64, (2 * DINNER) / 64), 128, 0, stream>>>(
      h1bf, DMODEL, w_in, nullptr, nullptr, xz, nullptr, NROWS, 2 * DINNER, DMODEL);

  // depthwise conv + silu
  conv_silu<<<(NROWS * DINNER) / 256, 256, 0, stream>>>(xz, conv_w, conv_b, xi, xibf);

  // x_proj: [8192,2048] x [96,2048]^T -> xdbl f32+bf16
  gemm_bf16<EP_F32_BF16><<<dim3(NROWS / 64, 2), 128, 0, stream>>>(
      xibf, DINNER, w_xp, nullptr, nullptr, xdbl, xdblb, NROWS, 96, DINNER);

  // dt_proj + softplus: [8192,64] (lda=96) x [2048,64]^T -> dt
  gemm_bf16<EP_SOFTPLUS><<<dim3(NROWS / 64, DINNER / 64), 128, 0, stream>>>(
      xdblb, 96, w_dt, dt_b, nullptr, dt_ys, nullptr, NROWS, DINNER, DTRANK);

  // sequential SSM scan (ys overwrites dt in place)
  ssm_scan<<<(BB * DINNER) / 256, 256, 0, stream>>>(xdbl, dt_ys, xi, A_log);

  // gate
  gate_silu<<<(NROWS * DINNER) / 256, 256, 0, stream>>>(dt_ys, xi, xz, Dp, ybf);

  // out_proj + residual: x2 = x + y @ W^T
  gemm_bf16<EP_ADDRES><<<dim3(NROWS / 64, DMODEL / 64), 128, 0, stream>>>(
      ybf, DINNER, w_op, nullptr, x, x2, nullptr, NROWS, DMODEL, DINNER);

  // LN2 -> bf16
  ln_bf16<<<NROWS, 256, 0, stream>>>(x2, ln2_g, ln2_b, h2bf);

  // FFN1 + bias + relu -> bf16
  gemm_bf16<EP_BIAS_RELU_BF16><<<dim3(NROWS / 64, (4 * DMODEL) / 64), 128, 0, stream>>>(
      h2bf, DMODEL, w_f1, ffn_b1, nullptr, nullptr, f1bf, NROWS, 4 * DMODEL, DMODEL);

  // FFN2 + bias + residual -> out
  gemm_bf16<EP_BIAS_ADD><<<dim3(NROWS / 64, DMODEL / 64), 128, 0, stream>>>(
      f1bf, 4 * DMODEL, w_f2, ffn_b2, x2, out, nullptr, NROWS, DMODEL, 4 * DMODEL);
}
